// STKNearestNeighborEntropyLoss_22024592294536
// MI455X (gfx1250) — matrix operand order repaired and checked by reading, not compiled, not run on hardware
//
#include <hip/hip_runtime.h>

// ---------------------------------------------------------------------------
// STK nearest-neighbor entropy loss, MI455X (gfx1250, wave32, WMMA).
// dists = student @ teacher^T  (8192x8192 from 8192x1024), top-5 per row,
// loss = -mean(log(mean(top5) + 1e-8)).
// f32 -> bf16 once, then bf16 WMMA GEMM (f32 accum) with fused per-strip
// top-5. LDS tiles are filled with gfx1250 async copies
// (global_load_async_to_lds_b128 / s_wait_asynccnt) -- no VGPR round-trip.
// ---------------------------------------------------------------------------

typedef __attribute__((ext_vector_type(16))) __bf16 v16bf;
typedef __attribute__((ext_vector_type(8)))  float  v8f;
typedef __attribute__((ext_vector_type(4)))  unsigned int v4u;

#define BDIM   8192            // rows (student) == cols (teacher rows)
#define DDIM   1024            // embedding dim (K)
#define KTILE  64              // K per LDS chunk (2 WMMA k-steps)
#define NCHUNK (DDIM / KTILE)  // 16 K-chunks
#define MT     128             // workgroup M tile
#define NT     128             // inner N tile
#define NSTRIP 512             // N strip per workgroup (4 inner tiles)
#define NINNER (NSTRIP / NT)
#define LDA    72              // LDS row stride in bf16 (64 + 8 pad, 144B = 16B aligned)
#define LDSTG  130             // staging row stride in floats (conflict-free)
#define NSTRIPS (BDIM / NSTRIP)  // 16 strips
#define NCAND  (NSTRIPS * 5)     // 80 candidates per row

// ---------------------------------------------------------------------------

__device__ __forceinline__ unsigned short f2bf(float x) {
  unsigned u = __float_as_uint(x);
  u += 0x7FFFu + ((u >> 16) & 1u);   // round-to-nearest-even
  return (unsigned short)(u >> 16);
}

__global__ __launch_bounds__(256) void cvt_kernel(const float* __restrict__ in,
                                                  unsigned short* __restrict__ out) {
  int i = blockIdx.x * 256 + threadIdx.x;     // one float4 per thread
  float4 f = ((const float4*)in)[i];
  unsigned lo = (unsigned)f2bf(f.x) | ((unsigned)f2bf(f.y) << 16);
  unsigned hi = (unsigned)f2bf(f.z) | ((unsigned)f2bf(f.w) << 16);
  ((uint2*)out)[i] = make_uint2(lo, hi);
}

// ---------------------------------------------------------------------------
// gfx1250 async copy: global memory -> LDS, 16B per lane, tracked by ASYNCcnt.

__device__ __forceinline__ void async_b128(unsigned lds_off, const unsigned short* g) {
  asm volatile("global_load_async_to_lds_b128 %0, %1, off"
               :: "v"(lds_off), "v"(g)
               : "memory");
}

__device__ __forceinline__ void wait_async0() {
#if __has_builtin(__builtin_amdgcn_s_wait_asynccnt)
  __builtin_amdgcn_s_wait_asynccnt(0);
#else
  asm volatile("s_wait_asynccnt 0x0" ::: "memory");
#endif
}

__device__ __forceinline__ unsigned lds_off32(const void* p) {
  // generic LDS pointer: low 32 bits are the wave-relative LDS byte address
  return (unsigned)(unsigned long long)p;
}

// ---------------------------------------------------------------------------

union Frag16 { v4u q[2]; v16bf v; };

extern __shared__ char smem_raw[];

__device__ __forceinline__ void top5_push(float v, float& t0, float& t1,
                                          float& t2, float& t3, float& t4) {
  if (v > t4) {
    t4 = v;
    if (t4 > t3) { float x = t3; t3 = t4; t4 = x; }
    if (t3 > t2) { float x = t2; t2 = t3; t3 = x; }
    if (t2 > t1) { float x = t1; t1 = t2; t2 = x; }
    if (t1 > t0) { float x = t0; t0 = t1; t1 = x; }
  }
}

__global__ __launch_bounds__(256) void gemm_topk_kernel(
    const unsigned short* __restrict__ sB,   // student bf16 [8192][1024]
    const unsigned short* __restrict__ tB,   // teacher bf16 [8192][1024]
    float* __restrict__ cand) {              // [8192][16][5]
  const int tid = threadIdx.x;
  const int bx  = blockIdx.x;   // row tile   (64)
  const int by  = blockIdx.y;   // col strip  (16)

  __bf16* As   = (__bf16*)smem_raw;            // 2 buffers * MT*LDA bf16
  __bf16* Bs   = As + 2 * MT * LDA;            // 2 buffers * MT*LDA bf16
  float*  stag = (float*)(Bs + 2 * MT * LDA);  // MT * LDSTG f32

  const int lane = tid & 31;
  const int wave = tid >> 5;    // 0..7
  const int wm   = wave & 3;    // 4 waves along M -> 32 rows each
  const int wn   = wave >> 2;   // 2 waves along N -> 64 cols each
  const int l15  = lane & 15;
  const int lhi  = lane >> 4;   // half-wave select

  // async fill mapping: chunk = 128 rows x 64 bf16 = 1024 16B-vectors per matrix;
  // thread t owns rows r0+32i (i=0..3) at bf16 column kf
  const int r0 = tid >> 3;          // 0..31
  const int kf = (tid & 7) * 8;     // 0..56

  const unsigned short* gA = sB + (size_t)bx * MT * DDIM;

  // running top-5, thread r (<128) owns output row bx*128 + r
  float t0 = -1e30f, t1 = -1e30f, t2 = -1e30f, t3 = -1e30f, t4 = -1e30f;

  for (int it = 0; it < NINNER; ++it) {
    const int n0 = by * NSTRIP + it * NT;
    const unsigned short* gB = tB + (size_t)n0 * DDIM;

    v8f acc[2][4];
    {
      v8f z = {};
#pragma unroll
      for (int mi = 0; mi < 2; ++mi)
#pragma unroll
        for (int ni = 0; ni < 4; ++ni) acc[mi][ni] = z;
    }

    // ---- prologue: async-fill chunk 0 into buffer 0
#pragma unroll
    for (int i = 0; i < 4; ++i) {
      const int r = r0 + 32 * i;
      async_b128(lds_off32(As + r * LDA + kf), gA + (size_t)r * DDIM + kf);
      async_b128(lds_off32(Bs + r * LDA + kf), gB + (size_t)r * DDIM + kf);
    }
    wait_async0();
    __syncthreads();

    // ---- K loop, double buffered via async copies
    for (int kc = 0; kc < NCHUNK; ++kc) {
      const int cur = kc & 1;
      const int nxt = cur ^ 1;

      if (kc + 1 < NCHUNK) {   // async-fill next chunk while computing
        const int kb = (kc + 1) * KTILE;
        __bf16* An = As + nxt * MT * LDA;
        __bf16* Bn = Bs + nxt * MT * LDA;
#pragma unroll
        for (int i = 0; i < 4; ++i) {
          const int r = r0 + 32 * i;
          async_b128(lds_off32(An + r * LDA + kf), gA + (size_t)r * DDIM + kb + kf);
          async_b128(lds_off32(Bn + r * LDA + kf), gB + (size_t)r * DDIM + kb + kf);
        }
      }

      const __bf16* Ab = As + cur * MT * LDA;
      const __bf16* Bb = Bs + cur * MT * LDA;
      const int ka  = lhi ? 8 : 0;    // A: lanes 0-15 K0..7/16..23, 16-31 K8..15/24..31
      const int kb2 = lhi ? 8 : 0;    // B: lanes 0-15 K0..7/16..23, 16-31 K8..15/24..31

#pragma unroll
      for (int ks = 0; ks < KTILE; ks += 32) {
        Frag16 fA[2];
#pragma unroll
        for (int mi = 0; mi < 2; ++mi) {
          const int row = wm * 32 + mi * 16 + l15;
          fA[mi].q[0] = *(const v4u*)(Ab + row * LDA + ks + ka);
          fA[mi].q[1] = *(const v4u*)(Ab + row * LDA + ks + 16 + ka);
        }
        Frag16 fB[4];
#pragma unroll
        for (int ni = 0; ni < 4; ++ni) {
          const int col = wn * 64 + ni * 16 + l15;
          fB[ni].q[0] = *(const v4u*)(Bb + col * LDA + ks + kb2);
          fB[ni].q[1] = *(const v4u*)(Bb + col * LDA + ks + kb2 + 16);
        }
#pragma unroll
        for (int mi = 0; mi < 2; ++mi)
#pragma unroll
          for (int ni = 0; ni < 4; ++ni)
            acc[mi][ni] = __builtin_amdgcn_wmma_f32_16x16x32_bf16(
                false, fA[mi].v, false, fB[ni].v, (short)0, acc[mi][ni],
                false, false);
      }

      if (kc + 1 < NCHUNK) {
        wait_async0();       // our async fill of buf[nxt] has landed
        __syncthreads();     // everyone's fill landed; prior reads retired
      }
    }

    // ---- spill accumulators to staging (C layout: VGPR j -> M=j+8*half, N=lane&15)
#pragma unroll
    for (int mi = 0; mi < 2; ++mi) {
#pragma unroll
      for (int ni = 0; ni < 4; ++ni) {
        const int col = wn * 64 + ni * 16 + l15;
#pragma unroll
        for (int j = 0; j < 8; ++j) {
          const int row = wm * 32 + mi * 16 + j + 8 * lhi;
          stag[row * LDSTG + col] = acc[mi][ni][j];
        }
      }
    }
    __syncthreads();

    // ---- per-row top-5 update over this 128x128 tile
    if (tid < MT) {
      const float* rp = stag + tid * LDSTG;
      for (int j = 0; j < NT; ++j) top5_push(rp[j], t0, t1, t2, t3, t4);
    }
    __syncthreads();
  }

  if (tid < MT) {
    const int row = bx * MT + tid;
    float* cp = cand + (size_t)row * NCAND + by * 5;
    cp[0] = t0; cp[1] = t1; cp[2] = t2; cp[3] = t3; cp[4] = t4;
  }
}

// ---------------------------------------------------------------------------

__global__ __launch_bounds__(256) void merge_kernel(const float* __restrict__ cand,
                                                    float* __restrict__ partial) {
  const int row = blockIdx.x * 256 + threadIdx.x;
  const float* cp = cand + (size_t)row * NCAND;
  float t0 = -1e30f, t1 = -1e30f, t2 = -1e30f, t3 = -1e30f, t4 = -1e30f;
  for (int i = 0; i < NCAND; ++i) top5_push(cp[i], t0, t1, t2, t3, t4);
  float mean = (t0 + t1 + t2 + t3 + t4) * 0.2f;
  float lv = logf(mean + 1e-8f);

  __shared__ float red[256];
  red[threadIdx.x] = lv;
  __syncthreads();
  for (int s = 128; s > 0; s >>= 1) {
    if (threadIdx.x < s) red[threadIdx.x] += red[threadIdx.x + s];
    __syncthreads();
  }
  if (threadIdx.x == 0) partial[blockIdx.x] = red[0];
}

__global__ void final_kernel(const float* __restrict__ partial,
                             float* __restrict__ out) {
  if (threadIdx.x == 0) {
    float s = 0.0f;
    for (int i = 0; i < 32; ++i) s += partial[i];
    out[0] = -s * (1.0f / (float)BDIM);
  }
}

// ---------------------------------------------------------------------------

extern "C" void kernel_launch(void* const* d_in, const int* in_sizes, int n_in,
                              void* d_out, int out_size, void* d_ws, size_t ws_size,
                              hipStream_t stream) {
  (void)in_sizes; (void)n_in; (void)out_size; (void)ws_size;

  const float* student = (const float*)d_in[0];
  const float* teacher = (const float*)d_in[1];
  float* out = (float*)d_out;

  char* ws = (char*)d_ws;
  unsigned short* sB = (unsigned short*)ws;                       // 16 MB
  unsigned short* tB = sB + (size_t)BDIM * DDIM;                  // 16 MB
  float* cand    = (float*)(ws + 2 * (size_t)BDIM * DDIM * sizeof(unsigned short));
  float* partial = cand + (size_t)BDIM * NCAND;                   // 32 floats

  const int n4 = BDIM * DDIM / 4;   // float4s per tensor
  cvt_kernel<<<n4 / 256, 256, 0, stream>>>(student, sB);
  cvt_kernel<<<n4 / 256, 256, 0, stream>>>(teacher, tB);

  const size_t smem = 2ull * 2ull * MT * LDA * sizeof(unsigned short)  // A+B dbl buf
                    + (size_t)MT * LDSTG * sizeof(float);              // staging
  gemm_topk_kernel<<<dim3(BDIM / MT, NSTRIPS), 256, smem, stream>>>(sB, tB, cand);

  merge_kernel<<<BDIM / 256, 256, 0, stream>>>(cand, partial);
  final_kernel<<<1, 32, 0, stream>>>(partial, out);
}
